// Char2Vec_27023934226760
// MI455X (gfx1250) — matrix operand order repaired and checked by reading, not compiled, not run on hardware
//
#include <hip/hip_runtime.h>
#include <stdint.h>

// ---------------------------------------------------------------------------
// Types for CDNA5 WMMA (wave32): v_wmma_f32_16x16x32_bf16
// ---------------------------------------------------------------------------
typedef __bf16 bf16_t;
typedef __attribute__((ext_vector_type(16))) __bf16 v16bf;
typedef __attribute__((ext_vector_type(8)))  float  v8f;

union Frag32 {
    v16bf    v;
    uint4    q[2];
    unsigned u[8];
};

__device__ __forceinline__ unsigned short f2bf(float f) {
    unsigned u = __float_as_uint(f);
    u += 0x7FFFu + ((u >> 16) & 1u);   // round-to-nearest-even
    return (unsigned short)(u >> 16);
}
__device__ __forceinline__ unsigned pack2bf(float lo, float hi) {
    return (unsigned)f2bf(lo) | ((unsigned)f2bf(hi) << 16);
}

// ---------------------------------------------------------------------------
// Problem constants
// ---------------------------------------------------------------------------
#define NWORDS 8192        // B*S
#define LCHARS 32
#define EMB    128
#define CH1    256
#define CH2    256

// LDS tile strides (bf16 elements); +8 pad => DWORD stride ≡ 4 (mod 64)
// giving per-lane bank rotation 4*t for conflict-free row access.
#define XS 136             // X rows: 34 (32 chars + 2 zero pad)
#define HS 264             // H1 rows: 34 (32 outputs + 2 zero pad)
#define WAVE_LDS_BYTES (34*XS*2 + 34*HS*2)   // 9248 + 17952 = 27200

// Workspace layout (bytes)
#define WS_P1   0                 // conv1 A-fragments: 12*16*32*16 bf16 = 196608 B
#define WS_P2   196608            // conv2 A-fragments: 24*16*32*16 bf16 = 393216 B
#define WS_PL   589824            // linear A-fragments: 8*16*32*16 bf16 = 131072 B
#define WS_M2B  720896            // pooled activations bf16: 8192*256*2 = 4 MiB
#define WS_M2F  4915200           // pooled activations f32 : 8192*256*4 = 8 MiB

// ---------------------------------------------------------------------------
// Kernel 1: convert weights to bf16 and pre-swizzle into per-lane WMMA
// A-fragment layout.  A-matrix (16x32, bf16) per-lane layout used here:
//   lane<16 : row M=lane,    elems = K kb+0..15 (contiguous)
//   lane>=16: row M=lane-16, elems = K kb+16..31 (contiguous)
// K ordering for convs: kk = k*Cin + ci  (so a 32-chunk stays within one tap)
// ---------------------------------------------------------------------------
__global__ __launch_bounds__(256)
void pack_weights(const float* __restrict__ w1, const float* __restrict__ w2,
                  const float* __restrict__ wl, char* __restrict__ ws) {
    int tid = blockIdx.x * blockDim.x + threadIdx.x;   // 22528 total
    const float* src; unsigned short* dst; int Cin, taps, local;
    if (tid < 6144)        { local = tid;         src = w1; dst = (unsigned short*)(ws + WS_P1); Cin = 128; taps = 3; }
    else if (tid < 18432)  { local = tid - 6144;  src = w2; dst = (unsigned short*)(ws + WS_P2); Cin = 256; taps = 3; }
    else                   { local = tid - 18432; src = wl; dst = (unsigned short*)(ws + WS_PL); Cin = 256; taps = 1; }

    int lane  = local & 31;
    int mtile = (local >> 5) & 15;
    int chunk = local >> 9;
    int co    = mtile * 16 + (lane & 15);
    int hi    = (lane >> 4) * 8;
    int kb    = chunk * 32;

    unsigned out[8];
#pragma unroll
    for (int p = 0; p < 8; ++p) {
        float f[2];
#pragma unroll
        for (int h = 0; h < 2; ++h) {
            int e  = 2 * p + h;
            int K  = kb + 2 * hi + e;
            int k  = K / Cin;
            int ci = K % Cin;
            f[h] = src[(co * Cin + ci) * taps + k];
        }
        out[p] = pack2bf(f[0], f[1]);
    }
    unsigned* dp = (unsigned*)(dst + (size_t)((chunk * 16 + mtile) * 32 + lane) * 16);
#pragma unroll
    for (int p = 0; p < 8; ++p) dp[p] = out[p];
}

// ---------------------------------------------------------------------------
// Kernel 2: per-wave fused embed -> conv1(+ReLU) -> conv2(+ReLU) -> maxpool.
// GEMM orientation: M = out-channel, N = time position, K = ci (x taps).
// B-operand (KxN) as used here (same k order as A): lane<16 holds K 0..15 of col N=lane; lane>=16
// holds K 16..31 of col N=lane-16  => one contiguous 32B LDS read per lane.
// ---------------------------------------------------------------------------
__global__ __launch_bounds__(64)
void char2vec_conv(const int* __restrict__ word, const float* __restrict__ emb,
                   const float* __restrict__ b1, const float* __restrict__ b2,
                   char* __restrict__ ws) {
    extern __shared__ char smem[];
    const int wave    = threadIdx.x >> 5;
    const int lane    = threadIdx.x & 31;
    const int ln      = lane & 15;          // column / row-within-tile index
    const int hi      = lane >> 4;          // 0/1 : K/row-half selector
    const int word_id = blockIdx.x * (blockDim.x >> 5) + wave;

    bf16_t*   X   = (bf16_t*)(smem + wave * WAVE_LDS_BYTES);
    bf16_t*   H1  = (bf16_t*)(smem + wave * WAVE_LDS_BYTES + 34 * XS * 2);
    unsigned* Xu  = (unsigned*)X;
    unsigned* H1u = (unsigned*)H1;

    const unsigned short* P1  = (const unsigned short*)(ws + WS_P1);
    const unsigned short* P2  = (const unsigned short*)(ws + WS_P2);
    unsigned*             M2B = (unsigned*)(ws + WS_M2B);
    float*                M2F = (float*)(ws + WS_M2F);

    // ---- zero the 2 pad rows of X and H1 (rows 32,33) ----
    for (int i = lane; i < 2 * (XS / 2); i += 32) Xu[32 * (XS / 2) + i] = 0u;
    for (int i = lane; i < 2 * (HS / 2); i += 32) H1u[32 * (HS / 2) + i] = 0u;

    // ---- embedding gather: lane = char position t, bf16-convert into X ----
    {
        int idx = word[word_id * LCHARS + lane];
        const float4* row = (const float4*)(emb + (size_t)idx * EMB);
#pragma unroll
        for (int j = 0; j < 32; ++j) {
            int jj = (j + lane) & 31;                  // stagger to spread banks
            float4 v = row[jj];
            Xu[lane * (XS / 2) + 2 * jj]     = pack2bf(v.x, v.y);
            Xu[lane * (XS / 2) + 2 * jj + 1] = pack2bf(v.z, v.w);
        }
    }

    // =======================  conv1: 128ci x 3taps -> 256co  ================
    // N-tiles: t in [0,16) (acc0) and [16,32) (acc1); t=30,31 are don't-care.
#pragma unroll 1
    for (int mtile = 0; mtile < 16; ++mtile) {
        v8f acc0 = {}, acc1 = {};
#pragma unroll
        for (int c = 0; c < 12; ++c) {                  // K chunks of 32
            int k  = c >> 2;
            int cb = (c & 3) << 5;
            Frag32 A, B0, B1;
            const uint4* ap = (const uint4*)(P1 + (size_t)((c * 16 + mtile) * 32 + lane) * 16);
            A.q[0] = ap[0]; A.q[1] = ap[1];
            const unsigned* bp0 = &Xu[(ln + k)      * (XS / 2) + (cb >> 1) + hi * 8];
            const unsigned* bp1 = &Xu[(16 + ln + k) * (XS / 2) + (cb >> 1) + hi * 8];
#pragma unroll
            for (int i = 0; i < 8; ++i) { B0.u[i] = bp0[i]; B1.u[i] = bp1[i]; }
            acc0 = __builtin_amdgcn_wmma_f32_16x16x32_bf16(false, A.v, false, B0.v, (short)0, acc0, false, false);
            acc1 = __builtin_amdgcn_wmma_f32_16x16x32_bf16(false, A.v, false, B1.v, (short)0, acc1, false, false);
        }
        // epilogue: +bias, ReLU, bf16-pack, store H1[t][co]
        int cobase = mtile * 16 + hi * 8;
        float4 bv0 = *(const float4*)(b1 + cobase);
        float4 bv1 = *(const float4*)(b1 + cobase + 4);
        float bb[8] = {bv0.x, bv0.y, bv0.z, bv0.w, bv1.x, bv1.y, bv1.z, bv1.w};
        unsigned* d0 = &H1u[ln        * (HS / 2) + (cobase >> 1)];
        unsigned* d1 = &H1u[(16 + ln) * (HS / 2) + (cobase >> 1)];
#pragma unroll
        for (int p = 0; p < 4; ++p) {
            d0[p] = pack2bf(fmaxf(acc0[2 * p] + bb[2 * p], 0.f),
                            fmaxf(acc0[2 * p + 1] + bb[2 * p + 1], 0.f));
            d1[p] = pack2bf(fmaxf(acc1[2 * p] + bb[2 * p], 0.f),
                            fmaxf(acc1[2 * p + 1] + bb[2 * p + 1], 0.f));
        }
    }

    // =======================  conv2: 256ci x 3taps -> 256co + maxpool  =====
    const float maskB = (ln < 12) ? 1.0f : 0.0f;        // t=16+ln valid iff t<=27
#pragma unroll 1
    for (int mtile = 0; mtile < 16; ++mtile) {
        v8f acc0 = {}, acc1 = {};
#pragma unroll
        for (int c = 0; c < 24; ++c) {                  // K chunks of 32 (K=768)
            int k  = c >> 3;
            int cb = (c & 7) << 5;
            Frag32 A, B0, B1;
            const uint4* ap = (const uint4*)(P2 + (size_t)((c * 16 + mtile) * 32 + lane) * 16);
            A.q[0] = ap[0]; A.q[1] = ap[1];
            const unsigned* bp0 = &H1u[(ln + k)      * (HS / 2) + (cb >> 1) + hi * 8];
            const unsigned* bp1 = &H1u[(16 + ln + k) * (HS / 2) + (cb >> 1) + hi * 8];
#pragma unroll
            for (int i = 0; i < 8; ++i) { B0.u[i] = bp0[i]; B1.u[i] = bp1[i]; }
            acc0 = __builtin_amdgcn_wmma_f32_16x16x32_bf16(false, A.v, false, B0.v, (short)0, acc0, false, false);
            acc1 = __builtin_amdgcn_wmma_f32_16x16x32_bf16(false, A.v, false, B1.v, (short)0, acc1, false, false);
        }
        int cobase = mtile * 16 + hi * 8;
        float4 bv0 = *(const float4*)(b2 + cobase);
        float4 bv1 = *(const float4*)(b2 + cobase + 4);
        float bb[8] = {bv0.x, bv0.y, bv0.z, bv0.w, bv1.x, bv1.y, bv1.z, bv1.w};
        float m2[8];
#pragma unroll
        for (int r = 0; r < 8; ++r) {
            float vA = fmaxf(acc0[r] + bb[r], 0.f);              // t = ln (0..15)
            float vB = fmaxf(acc1[r] + bb[r], 0.f) * maskB;      // t = 16+ln (<=27)
            m2[r] = fmaxf(vA, vB);                               // ReLU => max id = 0
        }
        // max over the 16 time-lanes (stays within each 16-lane half)
#pragma unroll
        for (int off = 1; off < 16; off <<= 1) {
#pragma unroll
            for (int r = 0; r < 8; ++r)
                m2[r] = fmaxf(m2[r], __shfl_xor(m2[r], off, 32));
        }
        if (ln == 0) {
            float*    mf = &M2F[(size_t)word_id * CH2 + cobase];
            unsigned* mb = &M2B[((size_t)word_id * CH2 + cobase) >> 1];
#pragma unroll
            for (int r = 0; r < 8; ++r) mf[r] = m2[r];
#pragma unroll
            for (int p = 0; p < 4; ++p) mb[p] = pack2bf(m2[2 * p], m2[2 * p + 1]);
        }
    }
}

// ---------------------------------------------------------------------------
// Kernel 3: out = relu(M2 @ Wl^T + bl) + M2.   One wave per 16 words.
// GEMM: M = co_out (16 tiles), N = word (16 per wave), K = 256 (8 chunks).
// ---------------------------------------------------------------------------
__global__ __launch_bounds__(128)
void linear_out(const float* __restrict__ bl, char* __restrict__ ws,
                float* __restrict__ out) {
    const int lane = threadIdx.x & 31;
    const int ln   = lane & 15;
    const int hi   = lane >> 4;
    const int wg   = blockIdx.x * (blockDim.x >> 5) + (threadIdx.x >> 5); // 0..511
    const int n0   = wg * 16;
    const int w    = n0 + ln;                         // this lane's word column

    const unsigned short* PL  = (const unsigned short*)(ws + WS_PL);
    const unsigned short* M2B = (const unsigned short*)(ws + WS_M2B);
    const float*          M2F = (const float*)(ws + WS_M2F);

#pragma unroll 1
    for (int mtile = 0; mtile < 16; ++mtile) {
        v8f acc = {};
#pragma unroll
        for (int c = 0; c < 8; ++c) {                  // K chunks of 32 (K=256)
            int cb = c << 5;
            Frag32 A, B;
            const uint4* ap = (const uint4*)(PL + (size_t)((c * 16 + mtile) * 32 + lane) * 16);
            A.q[0] = ap[0]; A.q[1] = ap[1];
            const uint4* bp = (const uint4*)(M2B + (size_t)w * CH2 + cb + hi * 16);
            B.q[0] = bp[0]; B.q[1] = bp[1];
            acc = __builtin_amdgcn_wmma_f32_16x16x32_bf16(false, A.v, false, B.v, (short)0, acc, false, false);
        }
        int cobase = mtile * 16 + hi * 8;
        float4 bv0 = *(const float4*)(bl + cobase);
        float4 bv1 = *(const float4*)(bl + cobase + 4);
        float bb[8] = {bv0.x, bv0.y, bv0.z, bv0.w, bv1.x, bv1.y, bv1.z, bv1.w};
        const float4* mf = (const float4*)(M2F + (size_t)w * CH2 + cobase);
        float4 m0 = mf[0], m1 = mf[1];
        float mm[8] = {m0.x, m0.y, m0.z, m0.w, m1.x, m1.y, m1.z, m1.w};
        float4 o0, o1;
        o0.x = fmaxf(acc[0] + bb[0], 0.f) + mm[0];
        o0.y = fmaxf(acc[1] + bb[1], 0.f) + mm[1];
        o0.z = fmaxf(acc[2] + bb[2], 0.f) + mm[2];
        o0.w = fmaxf(acc[3] + bb[3], 0.f) + mm[3];
        o1.x = fmaxf(acc[4] + bb[4], 0.f) + mm[4];
        o1.y = fmaxf(acc[5] + bb[5], 0.f) + mm[5];
        o1.z = fmaxf(acc[6] + bb[6], 0.f) + mm[6];
        o1.w = fmaxf(acc[7] + bb[7], 0.f) + mm[7];
        float4* op = (float4*)(out + (size_t)w * CH2 + cobase);
        op[0] = o0; op[1] = o1;
    }
}

// ---------------------------------------------------------------------------
extern "C" void kernel_launch(void* const* d_in, const int* in_sizes, int n_in,
                              void* d_out, int out_size, void* d_ws, size_t ws_size,
                              hipStream_t stream) {
    (void)in_sizes; (void)n_in; (void)out_size; (void)ws_size;
    const int*   word = (const int*)  d_in[0];
    const float* emb  = (const float*)d_in[1];
    const float* w1   = (const float*)d_in[2];
    const float* b1   = (const float*)d_in[3];
    const float* w2   = (const float*)d_in[4];
    const float* b2   = (const float*)d_in[5];
    const float* wl   = (const float*)d_in[6];
    const float* bl   = (const float*)d_in[7];
    float* out = (float*)d_out;
    char*  ws  = (char*)d_ws;

    // 1) bf16-convert + swizzle weights into WMMA A-fragment layout
    pack_weights<<<88, 256, 0, stream>>>(w1, w2, wl, ws);

    // 2) fused embed + conv1 + conv2 + maxpool: 1 wave/word, 2 waves/block
    char2vec_conv<<<NWORDS / 2, 64, 2 * WAVE_LDS_BYTES, stream>>>(word, emb, b1, b2, ws);

    // 3) linear + ReLU + residual: 1 wave per 16 words
    linear_out<<<NWORDS / 16 / 4, 128, 0, stream>>>(bl, ws, out);
}
